// highwayNet_g_compose_8538394984639
// MI455X (gfx1250) — compile-verified, matrix operand order checked by reading, not run on hardware
//
#include <hip/hip_runtime.h>
#include <hip/hip_bf16.h>
#include <math.h>

// ---------------------------------------------------------------------------
// HighwayNet-g compose for MI455X (gfx1250), wave32 + WMMA f16 (fp32 accum).
// T=30, N=4096, S=8, E=64, H=128, L=64.  Workspace need ~= 99 MB.
// ---------------------------------------------------------------------------

#define TT 30
#define NN 4096
#define HH 128
#define EE 64
#define LL 64
#define LN_EPS 1e-5f

typedef _Float16 half_t;
typedef __attribute__((ext_vector_type(16))) _Float16 v16h;
typedef __attribute__((ext_vector_type(8)))  float    v8f;

__device__ __forceinline__ v8f zero8() {
  v8f z = {0.f,0.f,0.f,0.f,0.f,0.f,0.f,0.f};
  return z;
}

__device__ __forceinline__ v8f wmma_f16(v16h a, v16h b, v8f c) {
  // D = A(16x32 f16) * B(32x16 f16) + C(16x16 f32)
  return __builtin_amdgcn_wmma_f32_16x16x32_f16(
      false, a, false, b, (short)0, c, false, false);
}

// Load a 16x32 f16 A/B fragment from a row-major tile (leading dim ld, elems).
// For A: "row" is the M row. For B from weight W[out][in]: same pattern with
// base = W + n0*K + k0, ld = K  (lane's low 4 bits select the output column).
__device__ __forceinline__ v16h load_ab_frag(const half_t* base, int ld) {
  int lane = threadIdx.x & 31;
  int r  = lane & 15;
  int hi = (lane >> 4) & 1;
  const half_t* p = base + (size_t)r * ld + hi * 8;
  union { v16h v; float4 f[2]; } u;
  u.f[0] = *(const float4*)(p);
  u.f[1] = *(const float4*)(p + 16);
  return u.v;
}

// Store a 16x16 fp32 C fragment to row-major dst (leading dim ld).
__device__ __forceinline__ void store_c(float* dst, int ld, v8f c) {
  int lane = threadIdx.x & 31;
  int col = lane & 15;
  int hi  = (lane >> 4) & 1;
#pragma unroll
  for (int i = 0; i < 8; ++i)
    dst[(size_t)(i + 8 * hi) * ld + col] = c[i];
}

__device__ __forceinline__ float wred32(float v) {
#pragma unroll
  for (int m = 16; m >= 1; m >>= 1) v += __shfl_xor(v, m, 32);
  return v;
}
__device__ __forceinline__ float wred16(float v) {
#pragma unroll
  for (int m = 8; m >= 1; m >>= 1) v += __shfl_xor(v, m, 16);
  return v;
}
__device__ __forceinline__ float leaky(float x) { return x > 0.f ? x : 0.1f * x; }
__device__ __forceinline__ float sigm(float x)  { return 1.f / (1.f + __expf(-x)); }

// ---------------------------------------------------------------------------
// fp32 -> f16 weight conversion
// ---------------------------------------------------------------------------
__global__ void f2h_kernel(const float* __restrict__ src, half_t* __restrict__ dst, int n) {
  int i = blockIdx.x * 256 + threadIdx.x;
  if (i < n) dst[i] = (half_t)src[i];
}

// ---------------------------------------------------------------------------
// cond embedding: [N,2] -> block(2,64) -> block(64,128) -> h0 [N,128] fp32
// one block (64 threads) per agent
// ---------------------------------------------------------------------------
__global__ __launch_bounds__(64) void cond_kernel(
    const float* __restrict__ cond,
    const float* __restrict__ w1, const float* __restrict__ b1,   // [64,2],[64]
    const float* __restrict__ w2, const float* __restrict__ b2,   // [128,64],[128]
    float* __restrict__ out)                                      // [N,128]
{
  __shared__ float e1[64];
  __shared__ float rs[64];
  __shared__ float rq[64];
  int n = blockIdx.x, tid = threadIdx.x;
  float c0 = cond[n * 2 + 0], c1 = cond[n * 2 + 1];
  float a = c0 * w1[tid * 2 + 0] + c1 * w1[tid * 2 + 1] + b1[tid];

  rs[tid] = a; rq[tid] = a * a; __syncthreads();
  for (int s = 32; s > 0; s >>= 1) {
    if (tid < s) { rs[tid] += rs[tid + s]; rq[tid] += rq[tid + s]; }
    __syncthreads();
  }
  float mean = rs[0] * (1.f / 64.f);
  float var  = rq[0] * (1.f / 64.f) - mean * mean;
  __syncthreads();
  float inv = rsqrtf(var + LN_EPS);
  e1[tid] = leaky((a - mean) * inv);
  __syncthreads();

  float oA = b2[tid], oB = b2[tid + 64];
  for (int k = 0; k < 64; ++k) {
    float e = e1[k];
    oA += e * w2[tid * 64 + k];
    oB += e * w2[(tid + 64) * 64 + k];
  }
  rs[tid] = oA + oB; rq[tid] = oA * oA + oB * oB; __syncthreads();
  for (int s = 32; s > 0; s >>= 1) {
    if (tid < s) { rs[tid] += rs[tid + s]; rq[tid] += rq[tid + s]; }
    __syncthreads();
  }
  mean = rs[0] * (1.f / 128.f);
  var  = rq[0] * (1.f / 128.f) - mean * mean;
  __syncthreads();
  inv = rsqrtf(var + LN_EPS);
  out[(size_t)n * 128 + tid]      = leaky((oA - mean) * inv);
  out[(size_t)n * 128 + tid + 64] = leaky((oB - mean) * inv);
}

// ---------------------------------------------------------------------------
// ip_emb_g: z [T*N,128] -> block(128,32) -> block(32,64) -> scene_emb f16
// one wave per row, 8 rows per block
// ---------------------------------------------------------------------------
__global__ __launch_bounds__(256) void ipemb_kernel(
    const float* __restrict__ z,
    const float* __restrict__ iw1, const float* __restrict__ ib1,  // [32,128]
    const float* __restrict__ iw2, const float* __restrict__ ib2,  // [64,32]
    half_t* __restrict__ semb)                                     // [T*N,64]
{
  __shared__ float aS[8][32];
  int wave = threadIdx.x >> 5, lane = threadIdx.x & 31;
  size_t r = (size_t)blockIdx.x * 8 + wave;   // < T*N
  const float* zr = z + r * 128;

  float a = ib1[lane];
  for (int k = 0; k < 128; ++k) a += zr[k] * iw1[lane * 128 + k];
  float sum = wred32(a), sq = wred32(a * a);
  float mean = sum * (1.f / 32.f), var = sq * (1.f / 32.f) - mean * mean;
  a = leaky((a - mean) * rsqrtf(var + LN_EPS));
  aS[wave][lane] = a;
  __syncthreads();

  float b0 = ib2[lane], b1 = ib2[lane + 32];
  for (int k = 0; k < 32; ++k) {
    float av = aS[wave][k];
    b0 += av * iw2[lane * 32 + k];
    b1 += av * iw2[(lane + 32) * 32 + k];
  }
  sum = wred32(b0 + b1); sq = wred32(b0 * b0 + b1 * b1);
  mean = sum * (1.f / 64.f); var = sq * (1.f / 64.f) - mean * mean;
  float inv = rsqrtf(var + LN_EPS);
  semb[r * 64 + lane]      = (half_t)leaky((b0 - mean) * inv);
  semb[r * 64 + lane + 32] = (half_t)leaky((b1 - mean) * inv);
}

// ---------------------------------------------------------------------------
// LayerNorm-GRU (persistent). Block owns 16 agents for all T steps.
// Fuses input projection (K=64) + recurrent projection (K=128) via WMMA.
// fwd (backward=0): enc[t] = h_t (f16)
// bwd (backward=1): enc[t] = 0.5*(h_t + enc_fwd[t]) in place
// ---------------------------------------------------------------------------
__global__ __launch_bounds__(256) void gru_kernel(
    const half_t* __restrict__ semb,   // [T*N,64] f16
    const half_t* __restrict__ wih,    // [384,64] f16
    const half_t* __restrict__ whh,    // [384,128] f16
    const float*  __restrict__ bih,    // [384]
    const float*  __restrict__ bhh,    // [384]
    const float*  __restrict__ h0,     // [N,128]
    half_t* __restrict__ enc,          // [T*N,128] f16
    int backward)
{
  __shared__ __align__(16) float  hS[16 * 128];   // fp32 hidden state
  __shared__ __align__(16) half_t hF[16 * 128];   // f16 copy for WMMA A
  __shared__ __align__(16) float  rzS[16 * 256];  // combined r|z pre-acts
  __shared__ __align__(16) float  hnS[16 * 128];  // recurrent part of n
  __shared__ __align__(16) float  inS[16 * 128];  // input part of n

  const int n0   = blockIdx.x * 16;
  const int tid  = threadIdx.x;
  const int wave = tid >> 5;

  for (int i = tid; i < 16 * 128; i += 256)
    hS[i] = h0[(size_t)(n0 + (i >> 7)) * 128 + (i & 127)];
  __syncthreads();

  for (int step = 0; step < TT; ++step) {
    const int t = backward ? (TT - 1 - step) : step;

    for (int i = tid; i < 16 * 128; i += 256) hF[i] = (half_t)hS[i];
    __syncthreads();

    // --- GEMM phase: wave w handles gate columns w*16..w*16+15 of each gate
    v8f c_r  = zero8();
    v8f c_z  = zero8();
    v8f c_nh = zero8();
    v8f c_ni = zero8();

#pragma unroll
    for (int kb = 0; kb < 4; ++kb) {            // recurrent: K = 128
      v16h a  = load_ab_frag(hF + kb * 32, 128);
      v16h br = load_ab_frag(whh + (size_t)(      wave * 16) * 128 + kb * 32, 128);
      v16h bz = load_ab_frag(whh + (size_t)(128 + wave * 16) * 128 + kb * 32, 128);
      v16h bn = load_ab_frag(whh + (size_t)(256 + wave * 16) * 128 + kb * 32, 128);
      c_r  = wmma_f16(a, br, c_r);
      c_z  = wmma_f16(a, bz, c_z);
      c_nh = wmma_f16(a, bn, c_nh);
    }
    const half_t* xbase = semb + (size_t)(t * NN + n0) * 64;
#pragma unroll
    for (int kb = 0; kb < 2; ++kb) {            // input: K = 64
      v16h a  = load_ab_frag(xbase + kb * 32, 64);
      v16h br = load_ab_frag(wih + (size_t)(      wave * 16) * 64 + kb * 32, 64);
      v16h bz = load_ab_frag(wih + (size_t)(128 + wave * 16) * 64 + kb * 32, 64);
      v16h bn = load_ab_frag(wih + (size_t)(256 + wave * 16) * 64 + kb * 32, 64);
      c_r  = wmma_f16(a, br, c_r);
      c_z  = wmma_f16(a, bz, c_z);
      c_ni = wmma_f16(a, bn, c_ni);
    }
    store_c(rzS +       wave * 16, 256, c_r);
    store_c(rzS + 128 + wave * 16, 256, c_z);
    store_c(hnS +       wave * 16, 128, c_nh);
    store_c(inS +       wave * 16, 128, c_ni);
    __syncthreads();

    // --- gate phase: thread -> (row = tid>>4, sublane s = tid&15), 8 ch each
    {
      const int row = tid >> 4;
      const int s   = tid & 15;
      float rv[8], zv[8], nv[8];
      float sum, sq, mean, var, inv;

      // r = sigmoid(LN(i_r + h_r))
      sum = 0.f; sq = 0.f;
#pragma unroll
      for (int j = 0; j < 8; ++j) {
        int c = s * 8 + j;
        float v = rzS[row * 256 + c] + bih[c] + bhh[c];
        rv[j] = v; sum += v; sq += v * v;
      }
      sum = wred16(sum); sq = wred16(sq);
      mean = sum * (1.f / 128.f); var = sq * (1.f / 128.f) - mean * mean;
      inv = rsqrtf(var + LN_EPS);
#pragma unroll
      for (int j = 0; j < 8; ++j) rv[j] = sigm((rv[j] - mean) * inv);

      // z = sigmoid(LN(i_z + h_z))
      sum = 0.f; sq = 0.f;
#pragma unroll
      for (int j = 0; j < 8; ++j) {
        int c = s * 8 + j;
        float v = rzS[row * 256 + 128 + c] + bih[128 + c] + bhh[128 + c];
        zv[j] = v; sum += v; sq += v * v;
      }
      sum = wred16(sum); sq = wred16(sq);
      mean = sum * (1.f / 128.f); var = sq * (1.f / 128.f) - mean * mean;
      inv = rsqrtf(var + LN_EPS);
#pragma unroll
      for (int j = 0; j < 8; ++j) zv[j] = sigm((zv[j] - mean) * inv);

      // n = tanh(LN(i_n + r * h_n))
      sum = 0.f; sq = 0.f;
#pragma unroll
      for (int j = 0; j < 8; ++j) {
        int c = s * 8 + j;
        float v = (inS[row * 128 + c] + bih[256 + c]) +
                  rv[j] * (hnS[row * 128 + c] + bhh[256 + c]);
        nv[j] = v; sum += v; sq += v * v;
      }
      sum = wred16(sum); sq = wred16(sq);
      mean = sum * (1.f / 128.f); var = sq * (1.f / 128.f) - mean * mean;
      inv = rsqrtf(var + LN_EPS);

#pragma unroll
      for (int j = 0; j < 8; ++j) {
        int c = s * 8 + j;
        float n_ = tanhf((nv[j] - mean) * inv);
        float ho = hS[row * 128 + c];
        float hn = (1.f - zv[j]) * n_ + zv[j] * ho;
        size_t oidx = ((size_t)t * NN + n0 + row) * 128 + c;
        if (!backward) {
          enc[oidx] = (half_t)hn;
        } else {
          float hf = (float)enc[oidx];
          enc[oidx] = (half_t)(0.5f * (hn + hf));
        }
        hS[row * 128 + c] = hn;
      }
    }
    __syncthreads();
  }
}

// ---------------------------------------------------------------------------
// seq = leaky(LN(hsum @ sw^T + sb)) : [T*N,128] x [128,64] via WMMA
// block = 256 thr / 8 waves, 128 rows per block
// ---------------------------------------------------------------------------
__global__ __launch_bounds__(256) void seq_kernel(
    const half_t* __restrict__ hsum,  // [T*N,128] f16 (avg of fwd/bwd enc)
    const half_t* __restrict__ swf,   // [64,128] f16
    const float*  __restrict__ sb,    // [64]
    half_t* __restrict__ seq)         // [T*N,64] f16
{
  __shared__ __align__(16) float oS[128 * 64];
  const int tid = threadIdx.x, wave = tid >> 5;
  const size_t r0 = (size_t)blockIdx.x * 128;

  v8f c0 = zero8(), c1 = zero8(), c2 = zero8(), c3 = zero8();
  const half_t* abase = hsum + (r0 + wave * 16) * 128;
#pragma unroll
  for (int kb = 0; kb < 4; ++kb) {
    v16h a  = load_ab_frag(abase + kb * 32, 128);
    v16h b0 = load_ab_frag(swf + (size_t)( 0) * 128 + kb * 32, 128);
    v16h b1 = load_ab_frag(swf + (size_t)(16) * 128 + kb * 32, 128);
    v16h b2 = load_ab_frag(swf + (size_t)(32) * 128 + kb * 32, 128);
    v16h b3 = load_ab_frag(swf + (size_t)(48) * 128 + kb * 32, 128);
    c0 = wmma_f16(a, b0, c0);
    c1 = wmma_f16(a, b1, c1);
    c2 = wmma_f16(a, b2, c2);
    c3 = wmma_f16(a, b3, c3);
  }
  store_c(oS + (size_t)(wave * 16) * 64 +  0, 64, c0);
  store_c(oS + (size_t)(wave * 16) * 64 + 16, 64, c1);
  store_c(oS + (size_t)(wave * 16) * 64 + 32, 64, c2);
  store_c(oS + (size_t)(wave * 16) * 64 + 48, 64, c3);
  __syncthreads();

  if (tid < 128) {
    float sum = 0.f, sq = 0.f;
    for (int c = 0; c < 64; ++c) {
      float v = oS[tid * 64 + c] + sb[c];
      sum += v; sq += v * v;
    }
    float mean = sum * (1.f / 64.f);
    float var  = sq * (1.f / 64.f) - mean * mean;
    float inv  = rsqrtf(var + LN_EPS);
    for (int c = 0; c < 64; ++c) {
      float v = oS[tid * 64 + c] + sb[c];
      seq[(r0 + tid) * 64 + c] = (half_t)leaky((v - mean) * inv);
    }
  }
}

// ---------------------------------------------------------------------------
// pairwise: z_rel = seq[src]-seq[dst]; block(mw,mb); mean over 7 pairs/agent
// grid (P/112, T); 7 waves; 112 pairs (16 agents) per block
// ---------------------------------------------------------------------------
__global__ __launch_bounds__(224) void rel_kernel(
    const half_t* __restrict__ seq,   // [T*N,64] f16
    const half_t* __restrict__ mwf,   // [64,64] f16
    const float*  __restrict__ mb,    // [64]
    const int*    __restrict__ src,
    const int*    __restrict__ dst,
    float* __restrict__ ave)          // [T*N,64] fp32
{
  __shared__ __align__(16) half_t aS[112 * 64];
  __shared__ __align__(16) float  oS[112 * 64];
  const int tid = threadIdx.x, wave = tid >> 5, lane = tid & 31;
  const int t = blockIdx.y;
  const int p0 = blockIdx.x * 112;
  const half_t* sq = seq + (size_t)t * NN * 64;

  // stage A tile: z_rel rows in f16
  for (int rr = 0; rr < 16; ++rr) {
    int p = p0 + wave * 16 + rr;
    int si = src[p], di = dst[p];
    int c = lane * 2;
    float v0 = (float)sq[(size_t)si * 64 + c]     - (float)sq[(size_t)di * 64 + c];
    float v1 = (float)sq[(size_t)si * 64 + c + 1] - (float)sq[(size_t)di * 64 + c + 1];
    aS[(size_t)(wave * 16 + rr) * 64 + c]     = (half_t)v0;
    aS[(size_t)(wave * 16 + rr) * 64 + c + 1] = (half_t)v1;
  }
  __syncthreads();

  v8f c0 = zero8(), c1 = zero8(), c2 = zero8(), c3 = zero8();
#pragma unroll
  for (int kb = 0; kb < 2; ++kb) {
    v16h a  = load_ab_frag(aS + (size_t)(wave * 16) * 64 + kb * 32, 64);
    v16h b0 = load_ab_frag(mwf + (size_t)( 0) * 64 + kb * 32, 64);
    v16h b1 = load_ab_frag(mwf + (size_t)(16) * 64 + kb * 32, 64);
    v16h b2 = load_ab_frag(mwf + (size_t)(32) * 64 + kb * 32, 64);
    v16h b3 = load_ab_frag(mwf + (size_t)(48) * 64 + kb * 32, 64);
    c0 = wmma_f16(a, b0, c0);
    c1 = wmma_f16(a, b1, c1);
    c2 = wmma_f16(a, b2, c2);
    c3 = wmma_f16(a, b3, c3);
  }
  store_c(oS + (size_t)(wave * 16) * 64 +  0, 64, c0);
  store_c(oS + (size_t)(wave * 16) * 64 + 16, 64, c1);
  store_c(oS + (size_t)(wave * 16) * 64 + 32, 64, c2);
  store_c(oS + (size_t)(wave * 16) * 64 + 48, 64, c3);
  __syncthreads();

  if (tid < 112) {  // LN + leaky per pair row
    float sum = 0.f, s2 = 0.f;
    for (int c = 0; c < 64; ++c) {
      float v = oS[tid * 64 + c] + mb[c];
      sum += v; s2 += v * v;
    }
    float mean = sum * (1.f / 64.f);
    float var  = s2 * (1.f / 64.f) - mean * mean;
    float inv  = rsqrtf(var + LN_EPS);
    for (int c = 0; c < 64; ++c) {
      float v = oS[tid * 64 + c] + mb[c];
      oS[tid * 64 + c] = leaky((v - mean) * inv);
    }
  }
  __syncthreads();

  // mean over the 7 pairs of each of the 16 agents
  for (int i = tid; i < 16 * 64; i += 224) {
    int a = i >> 6, c = i & 63;
    float s = 0.f;
    for (int k = 0; k < 7; ++k) s += oS[(size_t)(a * 7 + k) * 64 + c];
    ave[((size_t)t * NN + blockIdx.x * 16 + a) * 64 + c] = s * (1.f / 7.f);
  }
}

// ---------------------------------------------------------------------------
// head: full=[seq|ave](128) -> block(128,32) -> Linear(32,2) -> tanh
// one wave per row; out [N,2,T]
// ---------------------------------------------------------------------------
__global__ __launch_bounds__(256) void out_kernel(
    const half_t* __restrict__ seq,   // [T*N,64] f16
    const float*  __restrict__ ave,   // [T*N,64] fp32
    const float*  __restrict__ ow1, const float* __restrict__ ob1, // [32,128]
    const float*  __restrict__ ow2, const float* __restrict__ ob2, // [2,32]
    float* __restrict__ out)          // [N,2,T]
{
  __shared__ float fS[8][128];
  const int tid = threadIdx.x, wave = tid >> 5, lane = tid & 31;
  const size_t r = (size_t)blockIdx.x * 8 + wave;  // < T*N
  const int t = (int)(r / NN), n = (int)(r % NN);

  int c = lane * 2;
  fS[wave][c]          = (float)seq[r * 64 + c];
  fS[wave][c + 1]      = (float)seq[r * 64 + c + 1];
  fS[wave][64 + c]     = ave[r * 64 + c];
  fS[wave][64 + c + 1] = ave[r * 64 + c + 1];
  __syncthreads();

  float a1 = ob1[lane];
  for (int k = 0; k < 128; ++k) a1 += fS[wave][k] * ow1[lane * 128 + k];

  float sum = wred32(a1), sq = wred32(a1 * a1);
  float mean = sum * (1.f / 32.f), var = sq * (1.f / 32.f) - mean * mean;
  a1 = leaky((a1 - mean) * rsqrtf(var + LN_EPS));

  float s0 = wred32(a1 * ow2[lane]);
  float s1 = wred32(a1 * ow2[32 + lane]);
  if (lane < 2) {
    float v = tanhf((lane == 0 ? s0 : s1) + ob2[lane]);
    out[(size_t)n * (2 * TT) + lane * TT + t] = v;
  }
}

// ---------------------------------------------------------------------------
extern "C" void kernel_launch(void* const* d_in, const int* in_sizes, int n_in,
                              void* d_out, int out_size, void* d_ws, size_t ws_size,
                              hipStream_t stream) {
  (void)in_sizes; (void)n_in; (void)out_size; (void)ws_size;
  const float* cond  = (const float*)d_in[0];
  const float* z     = (const float*)d_in[1];
  const float* cw1   = (const float*)d_in[2];
  const float* cb1   = (const float*)d_in[3];
  const float* cw2   = (const float*)d_in[4];
  const float* cb2   = (const float*)d_in[5];
  const float* cbw1  = (const float*)d_in[6];
  const float* cbb1  = (const float*)d_in[7];
  const float* cbw2  = (const float*)d_in[8];
  const float* cbb2  = (const float*)d_in[9];
  const float* iw1   = (const float*)d_in[10];
  const float* ib1   = (const float*)d_in[11];
  const float* iw2   = (const float*)d_in[12];
  const float* ib2   = (const float*)d_in[13];
  const float* wih   = (const float*)d_in[14];
  const float* whh   = (const float*)d_in[15];
  const float* bih   = (const float*)d_in[16];
  const float* bhh   = (const float*)d_in[17];
  const float* wih_b = (const float*)d_in[18];
  const float* whh_b = (const float*)d_in[19];
  const float* bih_b = (const float*)d_in[20];
  const float* bhh_b = (const float*)d_in[21];
  const float* sw    = (const float*)d_in[22];
  const float* sb    = (const float*)d_in[23];
  const float* mw    = (const float*)d_in[24];
  const float* mb    = (const float*)d_in[25];
  const float* ow1   = (const float*)d_in[26];
  const float* ob1   = (const float*)d_in[27];
  const float* ow2   = (const float*)d_in[28];
  const float* ob2   = (const float*)d_in[29];
  const int* src_idx = (const int*)d_in[30];
  const int* dst_idx = (const int*)d_in[31];
  float* out = (float*)d_out;

  // workspace carve-out (256B aligned)
  char* ws = (char*)d_ws;
  size_t off = 0;
  auto carve = [&](size_t bytes) -> char* {
    char* p = ws + off;
    off += (bytes + 255) & ~(size_t)255;
    return p;
  };
  half_t* whh_f   = (half_t*)carve(384 * 128 * sizeof(half_t));
  half_t* wih_f   = (half_t*)carve(384 * 64  * sizeof(half_t));
  half_t* whh_bf  = (half_t*)carve(384 * 128 * sizeof(half_t));
  half_t* wih_bf  = (half_t*)carve(384 * 64  * sizeof(half_t));
  half_t* sw_f    = (half_t*)carve(64 * 128  * sizeof(half_t));
  half_t* mw_f    = (half_t*)carve(64 * 64   * sizeof(half_t));
  float*  disp    = (float*)carve((size_t)NN * 128 * sizeof(float));
  float*  disp_b  = (float*)carve((size_t)NN * 128 * sizeof(float));
  half_t* semb    = (half_t*)carve((size_t)TT * NN * 64  * sizeof(half_t));
  half_t* enc     = (half_t*)carve((size_t)TT * NN * 128 * sizeof(half_t));
  half_t* seqb    = (half_t*)carve((size_t)TT * NN * 64  * sizeof(half_t));
  float*  aveb    = (float*)carve((size_t)TT * NN * 64  * sizeof(float));

  const size_t ROWS = (size_t)TT * NN;  // 122880

  // 1) weights -> f16
  auto cvt = [&](const float* s, half_t* d, int n) {
    f2h_kernel<<<(n + 255) / 256, 256, 0, stream>>>(s, d, n);
  };
  cvt(whh,   whh_f,  384 * 128);
  cvt(wih,   wih_f,  384 * 64);
  cvt(whh_b, whh_bf, 384 * 128);
  cvt(wih_b, wih_bf, 384 * 64);
  cvt(sw,    sw_f,   64 * 128);
  cvt(mw,    mw_f,   64 * 64);

  // 2) condition embeddings -> h0 seeds
  cond_kernel<<<NN, 64, 0, stream>>>(cond, cw1, cb1, cw2, cb2, disp);
  cond_kernel<<<NN, 64, 0, stream>>>(cond, cbw1, cbb1, cbw2, cbb2, disp_b);

  // 3) input embedding -> scene_emb f16
  ipemb_kernel<<<(int)(ROWS / 8), 256, 0, stream>>>(z, iw1, ib1, iw2, ib2, semb);

  // 4) forward GRU (writes enc), 5) backward GRU (averages in place)
  gru_kernel<<<NN / 16, 256, 0, stream>>>(semb, wih_f, whh_f, bih, bhh,
                                          disp, enc, 0);
  gru_kernel<<<NN / 16, 256, 0, stream>>>(semb, wih_bf, whh_bf, bih_b, bhh_b,
                                          disp_b, enc, 1);

  // 6) seq projection H->L
  seq_kernel<<<(int)(ROWS / 128), 256, 0, stream>>>(enc, sw_f, sb, seqb);

  // 7) pairwise relative encodings + per-agent mean
  rel_kernel<<<dim3(NN / 16, TT), 224, 0, stream>>>(seqb, mw_f, mb,
                                                    src_idx, dst_idx, aveb);

  // 8) output head
  out_kernel<<<(int)(ROWS / 8), 256, 0, stream>>>(seqb, aveb, ow1, ob1,
                                                  ow2, ob2, out);
}
